// MultiplicativeAttention_2774548873320
// MI455X (gfx1250) — compile-verified, matrix operand order checked by reading, not run on hardware
//
#include <hip/hip_runtime.h>

typedef __attribute__((ext_vector_type(16))) _Float16 v16h;
typedef __attribute__((ext_vector_type(8)))  float    v8f;

#define WMMA_F16(A, Bm, C) \
  __builtin_amdgcn_wmma_f32_16x16x32_f16(false, (A), false, (Bm), (short)0, (C), false, false)

constexpr int cB = 2, cH = 16, cS = 2048, cD = 64;
constexpr int WAVES = 8;          // 256 threads per block
constexpr float SCALE = 0.125f;   // 1/sqrt(64)

// reduce across the 16-lane half (xor masks <16 never cross halves in wave32)
__device__ __forceinline__ float hmax16(float x) {
  x = fmaxf(x, __shfl_xor(x, 1, 32));
  x = fmaxf(x, __shfl_xor(x, 2, 32));
  x = fmaxf(x, __shfl_xor(x, 4, 32));
  x = fmaxf(x, __shfl_xor(x, 8, 32));
  return x;
}
__device__ __forceinline__ float hsum16(float x) {
  x += __shfl_xor(x, 1, 32);
  x += __shfl_xor(x, 2, 32);
  x += __shfl_xor(x, 4, 32);
  x += __shfl_xor(x, 8, 32);
  return x;
}

// Wave-local LDS ordering point: same-wave DS ops are in-order on CDNA5;
// this forces the compiler not to reorder and drains DScnt.
__device__ __forceinline__ void lds_fence_wave() {
  asm volatile("s_wait_dscnt 0x0" ::: "memory");
}

// Load a 16x32 (rows x depth) f16 WMMA A/B tile from a row-major f32 [*, 64] matrix.
// Lane (l&15) = row; lanes 0-15 hold depths {0..7,16..23}, lanes 16-31 hold {8..15,24..31}.
__device__ __forceinline__ v16h load_tile16x32(const float* __restrict__ base,
                                               int row0, int dchunk, int lane) {
  int r = row0 + (lane & 15);
  const float* p = base + (size_t)r * cD + dchunk * 32 + ((lane >> 4) & 1) * 8;
  const float4* p4 = (const float4*)p;
  float4 x0 = p4[0], x1 = p4[1];   // depths +0..7
  float4 x2 = p4[4], x3 = p4[5];   // depths +16..23
  v16h a;
  a[0] = (_Float16)x0.x;  a[1] = (_Float16)x0.y;  a[2]  = (_Float16)x0.z;  a[3]  = (_Float16)x0.w;
  a[4] = (_Float16)x1.x;  a[5] = (_Float16)x1.y;  a[6]  = (_Float16)x1.z;  a[7]  = (_Float16)x1.w;
  a[8] = (_Float16)x2.x;  a[9] = (_Float16)x2.y;  a[10] = (_Float16)x2.z;  a[11] = (_Float16)x2.w;
  a[12] = (_Float16)x3.x; a[13] = (_Float16)x3.y; a[14] = (_Float16)x3.z;  a[15] = (_Float16)x3.w;
  return a;
}

// Scaled+masked 16x32 score block (two 16x16 C tiles) for keys [n0, n0+32).
__device__ __forceinline__ void score_block(v16h aq0, v16h aq1,
                                            const float* __restrict__ kb,
                                            const int* __restrict__ mrow,
                                            int q0, int n0, int lane,
                                            v8f& clo, v8f& chi) {
  v16h b0 = load_tile16x32(kb, n0,      0, lane);
  v16h b1 = load_tile16x32(kb, n0,      1, lane);
  v16h b2 = load_tile16x32(kb, n0 + 16, 0, lane);
  v16h b3 = load_tile16x32(kb, n0 + 16, 1, lane);
  v8f c0 = {}; c0 = WMMA_F16(aq0, b0, c0); c0 = WMMA_F16(aq1, b1, c0);
  v8f c1 = {}; c1 = WMMA_F16(aq0, b2, c1); c1 = WMMA_F16(aq1, b3, c1);
  int col = lane & 15;
  int hi8 = ((lane >> 4) & 1) * 8;
#pragma unroll
  for (int r = 0; r < 8; ++r) {
    int row = q0 + r + hi8;                       // C layout: VGPR r -> M = r (+8 for hi lanes)
    int m0 = mrow[(size_t)row * cS + n0 + col];
    int m1 = mrow[(size_t)row * cS + n0 + 16 + col];
    c0[r] = (m0 == 1) ? -1e9f : c0[r] * SCALE;
    c1[r] = (m1 == 1) ? -1e9f : c1[r] * SCALE;
  }
  clo = c0; chi = c1;
}

__global__ __launch_bounds__(WAVES * 32)
void fused_attn_kernel(const float* __restrict__ q, const float* __restrict__ k,
                       const float* __restrict__ v, const int* __restrict__ mask,
                       float* __restrict__ res, float* __restrict__ attn) {
  __shared__ __align__(16) _Float16 plds[WAVES * 16 * 32];   // per-wave P transpose tile

  const int tid  = threadIdx.x;
  const int wave = tid >> 5;
  const int lane = tid & 31;
  const int tile = blockIdx.x * WAVES + wave;   // 4096 tiles total, exact fit
  const int bh   = tile >> 7;                   // tile / (S/16)
  const int q0   = (tile & 127) * 16;
  const int b    = bh >> 4;

  const size_t base = (size_t)bh * cS * cD;
  const float* qb = q + base;
  const float* kb = k + base;
  const float* vb = v + base;
  const int*   mrow  = mask + (size_t)b * cS * cS;
  float*       resb  = res  + base;
  float*       attnb = attn + (size_t)bh * cS * cS;

  const int col  = lane & 15;
  const int hsel = (lane >> 4) & 1;
  const int hi8  = hsel * 8;

  // Resident Q A-tiles (16 rows x 64 depth -> two 16x32 f16 tiles)
  v16h aq0 = load_tile16x32(qb, q0, 0, lane);
  v16h aq1 = load_tile16x32(qb, q0, 1, lane);

  // ---- Pass 1: online row max m and row sum l over all 2048 keys ----
  float m[8], l[8];
#pragma unroll
  for (int r = 0; r < 8; ++r) { m[r] = -__builtin_inff(); l[r] = 0.0f; }

  for (int j = 0; j < cS; j += 32) {
    if (j + 32 < cS) {  // prefetch next K block (32 rows x 256B) -> global_prefetch_b8
      const float* nk = kb + (size_t)(j + 32 + lane) * cD;
      __builtin_prefetch(nk, 0, 3);
      __builtin_prefetch(nk + 32, 0, 3);
    }
    v8f clo, chi;
    score_block(aq0, aq1, kb, mrow, q0, j, lane, clo, chi);
#pragma unroll
    for (int r = 0; r < 8; ++r) {
      float bm = hmax16(fmaxf(clo[r], chi[r]));
      float mn = fmaxf(m[r], bm);
      float s  = hsum16(__expf(clo[r] - mn) + __expf(chi[r] - mn));
      l[r] = l[r] * __expf(m[r] - mn) + s;
      m[r] = mn;
    }
  }
  float rl[8];
#pragma unroll
  for (int r = 0; r < 8; ++r) rl[r] = 1.0f / l[r];

  // ---- Pass 2: recompute scores, emit attn (streaming/NT), accumulate O = P @ V ----
  v8f o[4] = {v8f{}, v8f{}, v8f{}, v8f{}};
  _Float16* my = plds + wave * 16 * 32;

  for (int j = 0; j < cS; j += 32) {
    if (j + 32 < cS) {  // prefetch next K and V blocks
      const float* nk = kb + (size_t)(j + 32 + lane) * cD;
      const float* nv = vb + (size_t)(j + 32 + lane) * cD;
      __builtin_prefetch(nk, 0, 3);
      __builtin_prefetch(nk + 32, 0, 3);
      __builtin_prefetch(nv, 0, 3);
      __builtin_prefetch(nv + 32, 0, 3);
    }
    v8f clo, chi;
    score_block(aq0, aq1, kb, mrow, q0, j, lane, clo, chi);
#pragma unroll
    for (int r = 0; r < 8; ++r) {
      float e0 = __expf(clo[r] - m[r]);   // in [0,1]
      float e1 = __expf(chi[r] - m[r]);
      int row = q0 + r + hi8;
      // attn is a 512MB streaming output: bypass cache retention (NT hint)
      __builtin_nontemporal_store(e0 * rl[r], &attnb[(size_t)row * cS + j + col]);
      __builtin_nontemporal_store(e1 * rl[r], &attnb[(size_t)row * cS + j + 16 + col]);
      my[(r + hi8) * 32 + col]      = (_Float16)e0;   // C-layout -> LDS [row][key]
      my[(r + hi8) * 32 + 16 + col] = (_Float16)e1;
    }
    lds_fence_wave();  // wave-private tile: same-wave DS in-order, no workgroup barrier

    // Reload P as WMMA A-layout (16 rows x 32 keys)
    v16h p;
    {
      const _Float16* lp = my + (lane & 15) * 32 + hsel * 8;
#pragma unroll
      for (int i = 0; i < 8; ++i) p[i] = lp[i];          // keys hsel*8 + 0..7
#pragma unroll
      for (int i = 0; i < 8; ++i) p[8 + i] = lp[16 + i]; // keys hsel*8 + 16..23
    }

    // B = V tile (32 keys x 16 depth) per depth tile; lane = depth column
#pragma unroll
    for (int t = 0; t < 4; ++t) {
      const float* vp = vb + (size_t)(j + hi8) * cD + t * 16 + col;
      v16h bv;
#pragma unroll
      for (int i = 0; i < 8; ++i) bv[i]     = (_Float16)vp[(size_t)i * cD];
#pragma unroll
      for (int i = 0; i < 8; ++i) bv[8 + i] = (_Float16)vp[(size_t)(16 + i) * cD];
      o[t] = WMMA_F16(p, bv, o[t]);
    }
    asm volatile("" ::: "memory");  // keep next-iter LDS stores below this iter's loads
  }

  // ---- Normalize and write res (streaming output) ----
#pragma unroll
  for (int r = 0; r < 8; ++r) {
    int row = q0 + r + hi8;
    __builtin_nontemporal_store(o[0][r] * rl[r], &resb[(size_t)row * cD +  0 + col]);
    __builtin_nontemporal_store(o[1][r] * rl[r], &resb[(size_t)row * cD + 16 + col]);
    __builtin_nontemporal_store(o[2][r] * rl[r], &resb[(size_t)row * cD + 32 + col]);
    __builtin_nontemporal_store(o[3][r] * rl[r], &resb[(size_t)row * cD + 48 + col]);
  }
}

extern "C" void kernel_launch(void* const* d_in, const int* in_sizes, int n_in,
                              void* d_out, int out_size, void* d_ws, size_t ws_size,
                              hipStream_t stream) {
  const float* q    = (const float*)d_in[0];
  const float* k    = (const float*)d_in[1];
  const float* v    = (const float*)d_in[2];
  const int*   mask = (const int*)d_in[3];

  float* res  = (float*)d_out;                                 // [B,H,S,D]
  float* attn = res + (size_t)cB * cH * cS * cD;               // [B,H,S,S]

  const int tiles = cB * cH * (cS / 16);                       // 4096
  dim3 grid(tiles / WAVES), block(WAVES * 32);
  hipLaunchKernelGGL(fused_attn_kernel, grid, block, 0, stream,
                     q, k, v, mask, res, attn);
}
